// MyDeformNetwork_62397284876414
// MI455X (gfx1250) — compile-verified, matrix operand order checked by reading, not run on hardware
//
#include <hip/hip_runtime.h>
#include <cmath>

typedef __attribute__((ext_vector_type(16))) _Float16 v16h;
typedef __attribute__((ext_vector_type(8)))  float    v8f;

#define NPTS 200000
#define OUT_SCALE 600000
#define OUT_ROT   1200000

// ---- workspace layout (float offsets) ----
#define WS_VPOSED 0u          // N*3
#define WS_DELTA  600000u     // N*3
#define WS_LBSW   1200000u    // N*5
#define WS_JACC   2200000u    // 16
#define WS_ROT    2200016u    // 48 (5x9 used)
#define WS_PF     2200064u    // 48 (36 used)
#define WS_A      2200112u    // 80 (5x16)

// ---- padded LDS leading dimensions (halves) to break 256B bank periodicity ----
#define LD256 264
#define LD160 168
#define LD112 120
#define LD64  72
#define LD32  40
#define LD16  24

// ---- dynamic LDS layout: backbone kernel (bytes) ----
#define SM1_WBUF   0u          // 256 x LD256 f16 = 135168 B (layer1 uses LD32 stride)
#define SM1_BBUF   135168u     // 256 f32
#define SM1_ACTA   136192u     // 4 waves * 16*LD256 f16 = 33792 B
#define SM1_ACTB   169984u
#define SM1_EBUF   203776u     // 4 * 16*LD160 f16 = 21504 B
#define SM1_PBUF   225280u     // 4 * 16*LD112 f16 = 15360 B
#define SM1_DBUF   240640u     // 4 * 16*LD16 f16 = 3072 B
#define SM1_JRBUF  243712u
#define SM1_LBSBUF 246784u
#define SM1_BETAS  249856u     // 64 f32
#define SM1_PF     250112u     // 48 f32
#define SM1_JRED   250304u     // 16 f32
#define SM1_BYTES  250368u

// ---- dynamic LDS layout: front kernel (bytes) ----
#define SM3_WF0   0u         // 64 x LD32  = 5120 B
#define SM3_WF1   5120u      // 64 x LD64  = 9216 B
#define SM3_WF2   14336u
#define SM3_WS0   23552u
#define SM3_WS1   32768u
#define SM3_WS2   41984u     // 16 x LD64 = 2304 B
#define SM3_WR0   44288u
#define SM3_WR1   53504u
#define SM3_WR2   62720u     // 2304 B
#define SM3_BIAS  65024u     // 9 slots x 64 f32 = 2304 B
#define SM3_A     67328u     // 80 f32
#define SM3_FIN   67648u     // 4 * 16*LD32 f16 = 5120 B
#define SM3_FB1   72768u     // 4 * 16*LD64 f16 = 9216 B
#define SM3_FB2   81984u
#define SM3_FBF   91200u
#define SM3_SBUF  100416u    // 4 * 16*LD16 f16 = 3072 B
#define SM3_RBUF  103488u
#define SM3_BYTES 106560u

// ======================= WMMA fragment helpers =======================
// K-index map for 16-bit A/B fragments (ISA 7.12.2): lane-half h selects
// K 0-7/16-23 (h=0) vs 8-15/24-31 (h=1); element i picks the VGPR pair i/2.
__device__ __forceinline__ int kmap(int i, int h) {
  return ((i & 8) << 1) + (i & 7) + (h << 3);
}

__device__ __forceinline__ v16h frag_a(const _Float16* act, int lane, int k0, int ld) {
  const int m = lane & 15, h = lane >> 4;
  v16h a;
#pragma unroll
  for (int i = 0; i < 16; ++i) a[i] = act[m * ld + k0 + kmap(i, h)];
  return a;
}

__device__ __forceinline__ v16h frag_b(const _Float16* w, int lane, int n0, int k0, int ld) {
  const int n = lane & 15, h = lane >> 4;
  v16h b;
#pragma unroll
  for (int i = 0; i < 16; ++i) b[i] = w[(n0 + n) * ld + k0 + kmap(i, h)];
  return b;
}

// NT independent 16x16 accumulators share each A fragment: 4 parallel WMMA
// dependency chains per K-step (hides XDL latency at 1 wave/SIMD) and 1/NT
// the LDS traffic on A.
template <int NT>
__device__ __forceinline__ void wave_gemm_tiles(const _Float16* act, int lda,
                                                const _Float16* w, int ldw,
                                                const float* bl, _Float16* out, int ldo,
                                                int n0base, int Kin, int lane, bool relu) {
  const int n = lane & 15, h = lane >> 4;
  v8f c[NT];
#pragma unroll
  for (int t = 0; t < NT; ++t)
#pragma unroll
    for (int r = 0; r < 8; ++r) c[t][r] = 0.f;
  for (int k0 = 0; k0 < Kin; k0 += 32) {
    v16h a = frag_a(act, lane, k0, lda);
#pragma unroll
    for (int t = 0; t < NT; ++t) {
      v16h b = frag_b(w, lane, n0base + 16 * t, k0, ldw);
      c[t] = __builtin_amdgcn_wmma_f32_16x16x32_f16(false, a, false, b,
                                                    (short)0, c[t], false, false);
    }
  }
#pragma unroll
  for (int t = 0; t < NT; ++t) {
    const float bias = bl[n0base + 16 * t + n];
#pragma unroll
    for (int r = 0; r < 8; ++r) {
      const int m = r + 8 * h;             // C/D layout: M = vgpr + 8*half
      float v = c[t][r] + bias;
      if (relu) v = v > 0.f ? v : 0.f;
      out[m * ldo + n0base + 16 * t + n] = (_Float16)v;
    }
  }
}

__device__ __forceinline__ void wave_gemm(const _Float16* act, int lda,
                                          const _Float16* w, int ldw, const float* bl,
                                          _Float16* out, int ldo,
                                          int Kin, int doutPad, int lane, bool relu) {
  int tiles = doutPad >> 4, n0 = 0;
  while (tiles >= 4) {
    wave_gemm_tiles<4>(act, lda, w, ldw, bl, out, ldo, n0, Kin, lane, relu);
    n0 += 64; tiles -= 4;
  }
  if (tiles == 3)      wave_gemm_tiles<3>(act, lda, w, ldw, bl, out, ldo, n0, Kin, lane, relu);
  else if (tiles == 2) wave_gemm_tiles<2>(act, lda, w, ldw, bl, out, ldo, n0, Kin, lane, relu);
  else if (tiles == 1) wave_gemm_tiles<1>(act, lda, w, ldw, bl, out, ldo, n0, Kin, lane, relu);
}

// Stage f32 weights [dout][din] into LDS f16 [doutPad][ldw] (zero padded to Kpad)
__device__ __forceinline__ void stage_wb(_Float16* wl, float* bl,
                                         const float* W, const float* B,
                                         int dout, int din, int Kpad, int doutPad,
                                         int ldw, int tid, int nthr) {
  const int total = doutPad * Kpad;
  for (int idx = tid; idx < total; idx += nthr) {
    const int o = idx / Kpad, k = idx % Kpad;
    wl[o * ldw + k] = (_Float16)((o < dout && k < din) ? W[o * din + k] : 0.f);
  }
  for (int o = tid; o < doutPad; o += nthr) bl[o] = (o < dout) ? B[o] : 0.f;
}

// ======================= parameter bundles =======================
struct BackboneParams {
  const float *W1, *b1, *W2, *b2, *W3, *b3;
  const float *Wj0, *bj0, *Wj1, *bj1;
  const float *Wl0, *bl0, *Wl1, *bl1;
  const float *We, *be, *Wp, *bp, *Wd, *bd;
};
struct FrontParams {
  const float *Wf0, *bf0, *Wf1, *bf1, *Wf2, *bf2;
  const float *Ws0, *bs0, *Ws1, *bs1, *Ws2, *bs2;
  const float *Wr0, *br0, *Wr1, *br1, *Wr2, *br2;
};

// ======================= kernel 0: pose prep =======================
__global__ void deform_prep_kernel(const float* __restrict__ pose, float* ws) {
  const int tid = threadIdx.x;
  if (tid < 16) ws[WS_JACC + tid] = 0.f;
  if (tid == 0) {
    float* rot = ws + WS_ROT;
    float* pf  = ws + WS_PF;
    for (int j = 0; j < 5; ++j) {
      const float vx = pose[j * 3 + 0], vy = pose[j * 3 + 1], vz = pose[j * 3 + 2];
      const float ax = vx + 1e-8f, ay = vy + 1e-8f, az = vz + 1e-8f;
      const float ang = sqrtf(ax * ax + ay * ay + az * az);
      const float rx = vx / ang, ry = vy / ang, rz = vz / ang;
      const float c = cosf(ang), s = sinf(ang);
      const float K[9] = {0.f, -rz, ry, rz, 0.f, -rx, -ry, rx, 0.f};
      float K2[9];
      for (int i = 0; i < 3; ++i)
        for (int jj = 0; jj < 3; ++jj) {
          float acc = 0.f;
          for (int q = 0; q < 3; ++q) acc += K[i * 3 + q] * K[q * 3 + jj];
          K2[i * 3 + jj] = acc;
        }
      for (int i = 0; i < 9; ++i) {
        const float e = ((i % 4) == 0) ? 1.f : 0.f;
        rot[j * 9 + i] = e + s * K[i] + (1.f - c) * K2[i];
      }
    }
    for (int j = 1; j < 5; ++j)
      for (int i = 0; i < 9; ++i) {
        const float e = ((i % 4) == 0) ? 1.f : 0.f;
        pf[(j - 1) * 9 + i] = rot[j * 9 + i] - e;
      }
  }
}

// ======================= kernel 1: backbone (WMMA) =======================
__global__ void __launch_bounds__(128, 1)
deform_backbone_kernel(const float* __restrict__ x, const float* __restrict__ betas,
                       BackboneParams p, float* ws) {
  extern __shared__ char smem[];
  _Float16* wbuf    = (_Float16*)(smem + SM1_WBUF);
  float*    bbuf    = (float*)   (smem + SM1_BBUF);
  _Float16* actA    = (_Float16*)(smem + SM1_ACTA);
  _Float16* actB    = (_Float16*)(smem + SM1_ACTB);
  _Float16* ebuf    = (_Float16*)(smem + SM1_EBUF);
  _Float16* pbuf    = (_Float16*)(smem + SM1_PBUF);
  _Float16* dbuf    = (_Float16*)(smem + SM1_DBUF);
  _Float16* jrbuf   = (_Float16*)(smem + SM1_JRBUF);
  _Float16* lbsbuf  = (_Float16*)(smem + SM1_LBSBUF);
  float*    betas_s = (float*)(smem + SM1_BETAS);
  float*    pf_s    = (float*)(smem + SM1_PF);
  float*    Jred    = (float*)(smem + SM1_JRED);

  const int tid = threadIdx.x, nthr = blockDim.x;
  const int wave = tid >> 5, lane = tid & 31;
  const int p0 = blockIdx.x * 64 + wave * 16;

  _Float16* aw = actA + wave * (16 * LD256);
  _Float16* bw = actB + wave * (16 * LD256);
  _Float16* ew = ebuf + wave * (16 * LD160);
  _Float16* pw = pbuf + wave * (16 * LD112);
  _Float16* dw = dbuf  + wave * (16 * LD16);
  _Float16* jw = jrbuf + wave * (16 * LD16);
  _Float16* lw = lbsbuf + wave * (16 * LD16);

  // constants + reduction init
  for (int i = tid; i < 50; i += nthr) betas_s[i] = betas[i];
  for (int i = tid; i < 36; i += nthr) pf_s[i] = ws[WS_PF + i];
  if (tid < 16) Jred[tid] = 0.f;
  if (tid == 0) __builtin_prefetch(p.W2, 0, 1);   // global_prefetch_b8: warm L2

  // x tile -> A fragments [16][LD32] (K padded with zeros up to 32)
  for (int idx = lane; idx < 512; idx += 32) {
    const int m = idx >> 5, k = idx & 31;
    aw[m * LD32 + k] = (_Float16)((k < 3) ? x[(p0 + m) * 3 + k] : 0.f);
  }

  // L1: 3->256
  __syncthreads();
  stage_wb(wbuf, bbuf, p.W1, p.b1, 256, 3, 32, 256, LD32, tid, nthr);
  __syncthreads();
  wave_gemm(aw, LD32, wbuf, LD32, bbuf, bw, LD256, 32, 256, lane, true);
  // L2: 256->256
  __syncthreads();
  stage_wb(wbuf, bbuf, p.W2, p.b2, 256, 256, 256, 256, LD256, tid, nthr);
  __syncthreads();
  wave_gemm(bw, LD256, wbuf, LD256, bbuf, aw, LD256, 256, 256, lane, true);
  // L3: 256->256 ; h lives in bw afterwards
  __syncthreads();
  stage_wb(wbuf, bbuf, p.W3, p.b3, 256, 256, 256, 256, LD256, tid, nthr);
  __syncthreads();
  wave_gemm(aw, LD256, wbuf, LD256, bbuf, bw, LD256, 256, 256, lane, true);
  // jr head: 256->256 -> 5
  __syncthreads();
  stage_wb(wbuf, bbuf, p.Wj0, p.bj0, 256, 256, 256, 256, LD256, tid, nthr);
  __syncthreads();
  wave_gemm(bw, LD256, wbuf, LD256, bbuf, aw, LD256, 256, 256, lane, true);
  __syncthreads();
  stage_wb(wbuf, bbuf, p.Wj1, p.bj1, 5, 256, 256, 16, LD256, tid, nthr);
  __syncthreads();
  wave_gemm(aw, LD256, wbuf, LD256, bbuf, jw, LD16, 256, 16, lane, true);
  // lbsW head: 256->256 -> 5
  __syncthreads();
  stage_wb(wbuf, bbuf, p.Wl0, p.bl0, 256, 256, 256, 256, LD256, tid, nthr);
  __syncthreads();
  wave_gemm(bw, LD256, wbuf, LD256, bbuf, aw, LD256, 256, 256, lane, true);
  __syncthreads();
  stage_wb(wbuf, bbuf, p.Wl1, p.bl1, 5, 256, 256, 16, LD256, tid, nthr);
  __syncthreads();
  wave_gemm(aw, LD256, wbuf, LD256, bbuf, lw, LD16, 256, 16, lane, true);
  // exp basis: 256->150
  __syncthreads();
  stage_wb(wbuf, bbuf, p.We, p.be, 150, 256, 256, 160, LD256, tid, nthr);
  __syncthreads();
  wave_gemm(bw, LD256, wbuf, LD256, bbuf, ew, LD160, 256, 160, lane, false);
  // pose basis: 256->108
  __syncthreads();
  stage_wb(wbuf, bbuf, p.Wp, p.bp, 108, 256, 256, 112, LD256, tid, nthr);
  __syncthreads();
  wave_gemm(bw, LD256, wbuf, LD256, bbuf, pw, LD112, 256, 112, lane, false);
  // delta xyz: 256->3
  __syncthreads();
  stage_wb(wbuf, bbuf, p.Wd, p.bd, 3, 256, 256, 16, LD256, tid, nthr);
  __syncthreads();
  wave_gemm(bw, LD256, wbuf, LD256, bbuf, dw, LD16, 256, 16, lane, false);
  __syncthreads();

  // ---- per-point epilogue: lanes 0..15 each own one point ----
  if (lane < 16) {
    const int m = lane, pt = p0 + m;
    float d[3], vsh[3], po[3];
#pragma unroll
    for (int k = 0; k < 3; ++k) d[k] = (float)dw[m * LD16 + k];
#pragma unroll
    for (int k = 0; k < 3; ++k) {
      float acc = 0.f;
      for (int l = 0; l < 50; ++l)
        acc += (float)ew[m * LD160 + k * 50 + l] * betas_s[l];
      vsh[k] = x[pt * 3 + k] + d[k] + acc;
    }
#pragma unroll
    for (int k = 0; k < 3; ++k) {
      float acc = 0.f;
      for (int r = 0; r < 36; ++r)
        acc += pf_s[r] * (float)pw[m * LD112 + r * 3 + k];
      po[k] = acc;
    }
#pragma unroll
    for (int k = 0; k < 3; ++k) {
      ws[WS_VPOSED + (size_t)pt * 3 + k] = vsh[k] + po[k];
      ws[WS_DELTA  + (size_t)pt * 3 + k] = d[k];
    }
    // softmax over 5 joints
    float raw[5], mx = -1e30f, sum = 0.f;
    for (int j = 0; j < 5; ++j) { raw[j] = (float)lw[m * LD16 + j]; mx = fmaxf(mx, raw[j]); }
    for (int j = 0; j < 5; ++j) { raw[j] = expf(raw[j] - mx); sum += raw[j]; }
    for (int j = 0; j < 5; ++j) ws[WS_LBSW + (size_t)pt * 5 + j] = raw[j] / sum;
    // joint regression partial: J[j] += jr[j] * v_shaped
    for (int j = 0; j < 5; ++j) {
      const float jr = (float)jw[m * LD16 + j];
      for (int k = 0; k < 3; ++k) atomicAdd(&Jred[j * 3 + k], jr * vsh[k]);
    }
  }
  __syncthreads();
  if (tid < 15) atomicAdd(&ws[WS_JACC + tid], Jred[tid]);
}

// ======================= kernel 2: rigid transforms =======================
__device__ __forceinline__ void mat4mul(const float* a, const float* b, float* c) {
  for (int i = 0; i < 4; ++i)
    for (int j = 0; j < 4; ++j) {
      float acc = 0.f;
      for (int q = 0; q < 4; ++q) acc += a[i * 4 + q] * b[q * 4 + j];
      c[i * 4 + j] = acc;
    }
}

__global__ void deform_joints_kernel(float* ws) {
  if (threadIdx.x != 0) return;
  const int par[5] = {-1, 0, 1, 1, 1};
  float J[5][3];
  for (int j = 0; j < 5; ++j)
    for (int k = 0; k < 3; ++k) J[j][k] = ws[WS_JACC + j * 3 + k];
  float rel[5][3];
  for (int k = 0; k < 3; ++k) rel[0][k] = J[0][k];
  for (int j = 1; j < 5; ++j)
    for (int k = 0; k < 3; ++k) rel[j][k] = J[j][k] - J[par[j]][k];
  float Tm[5][16], W[5][16];
  for (int j = 0; j < 5; ++j) {
    for (int i = 0; i < 3; ++i) {
      for (int c = 0; c < 3; ++c) Tm[j][i * 4 + c] = ws[WS_ROT + j * 9 + i * 3 + c];
      Tm[j][i * 4 + 3] = rel[j][i];
    }
    Tm[j][12] = 0.f; Tm[j][13] = 0.f; Tm[j][14] = 0.f; Tm[j][15] = 1.f;
  }
  for (int q = 0; q < 16; ++q) W[0][q] = Tm[0][q];
  for (int j = 1; j < 5; ++j) mat4mul(W[par[j]], Tm[j], W[j]);
  for (int j = 0; j < 5; ++j) {
    float corr[3];
    for (int i = 0; i < 3; ++i) {
      float acc = 0.f;
      for (int q = 0; q < 3; ++q) acc += W[j][i * 4 + q] * J[j][q];
      corr[i] = acc;
    }
    for (int q = 0; q < 16; ++q) ws[WS_A + j * 16 + q] = W[j][q];
    for (int i = 0; i < 3; ++i) ws[WS_A + j * 16 + i * 4 + 3] = W[j][i * 4 + 3] - corr[i];
  }
}

// ======================= kernel 3: LBS + front MLP (WMMA) =======================
__global__ void __launch_bounds__(128, 1)
deform_front_kernel(const float* __restrict__ x, FrontParams p,
                    const float* __restrict__ ws, float* __restrict__ out) {
  extern __shared__ char smem[];
  _Float16* wf0 = (_Float16*)(smem + SM3_WF0);
  _Float16* wf1 = (_Float16*)(smem + SM3_WF1);
  _Float16* wf2 = (_Float16*)(smem + SM3_WF2);
  _Float16* wS0 = (_Float16*)(smem + SM3_WS0);
  _Float16* wS1 = (_Float16*)(smem + SM3_WS1);
  _Float16* wS2 = (_Float16*)(smem + SM3_WS2);
  _Float16* wR0 = (_Float16*)(smem + SM3_WR0);
  _Float16* wR1 = (_Float16*)(smem + SM3_WR1);
  _Float16* wR2 = (_Float16*)(smem + SM3_WR2);
  float*    bias = (float*)(smem + SM3_BIAS);
  float*    A_s  = (float*)(smem + SM3_A);
  _Float16* fin  = (_Float16*)(smem + SM3_FIN);
  _Float16* fb1  = (_Float16*)(smem + SM3_FB1);
  _Float16* fb2  = (_Float16*)(smem + SM3_FB2);
  _Float16* fbF  = (_Float16*)(smem + SM3_FBF);
  _Float16* sbuf = (_Float16*)(smem + SM3_SBUF);
  _Float16* rbuf = (_Float16*)(smem + SM3_RBUF);

  const int tid = threadIdx.x, nthr = blockDim.x;
  const int wave = tid >> 5, lane = tid & 31;
  const int p0 = blockIdx.x * 64 + wave * 16;

  // stage all weights once (they are tiny and L2-resident)
  stage_wb(wf0, bias + 0 * 64, p.Wf0, p.bf0, 64, 9, 32, 64, LD32, tid, nthr);
  stage_wb(wf1, bias + 1 * 64, p.Wf1, p.bf1, 64, 64, 64, 64, LD64, tid, nthr);
  stage_wb(wf2, bias + 2 * 64, p.Wf2, p.bf2, 64, 64, 64, 64, LD64, tid, nthr);
  stage_wb(wS0, bias + 3 * 64, p.Ws0, p.bs0, 64, 64, 64, 64, LD64, tid, nthr);
  stage_wb(wS1, bias + 4 * 64, p.Ws1, p.bs1, 64, 64, 64, 64, LD64, tid, nthr);
  stage_wb(wS2, bias + 5 * 64, p.Ws2, p.bs2, 3, 64, 64, 16, LD64, tid, nthr);
  stage_wb(wR0, bias + 6 * 64, p.Wr0, p.br0, 64, 64, 64, 64, LD64, tid, nthr);
  stage_wb(wR1, bias + 7 * 64, p.Wr1, p.br1, 64, 64, 64, 64, LD64, tid, nthr);
  stage_wb(wR2, bias + 8 * 64, p.Wr2, p.br2, 4, 64, 64, 16, LD64, tid, nthr);
  for (int i = tid; i < 80; i += nthr) A_s[i] = ws[WS_A + i];

  _Float16* fiw = fin + wave * (16 * LD32);
  _Float16* f1w = fb1 + wave * (16 * LD64);
  _Float16* f2w = fb2 + wave * (16 * LD64);
  _Float16* fFw = fbF + wave * (16 * LD64);
  _Float16* sw  = sbuf + wave * (16 * LD16);
  _Float16* rw  = rbuf + wave * (16 * LD16);

  for (int idx = lane; idx < 16 * LD32; idx += 32) fiw[idx] = (_Float16)0.f;
  __syncthreads();

  // ---- LBS per point, write xyz output and the 9-wide front input ----
  if (lane < 16) {
    const int m = lane, pt = p0 + m;
    float w5[5];
    for (int j = 0; j < 5; ++j) w5[j] = ws[WS_LBSW + (size_t)pt * 5 + j];
    float T[16];
#pragma unroll
    for (int q = 0; q < 16; ++q) {
      float acc = 0.f;
      for (int j = 0; j < 5; ++j) acc += w5[j] * A_s[j * 16 + q];
      T[q] = acc;
    }
    float vh[4] = {ws[WS_VPOSED + (size_t)pt * 3 + 0],
                   ws[WS_VPOSED + (size_t)pt * 3 + 1],
                   ws[WS_VPOSED + (size_t)pt * 3 + 2], 1.f};
#pragma unroll
    for (int k = 0; k < 3; ++k) {
      const float vert = T[k * 4 + 0] * vh[0] + T[k * 4 + 1] * vh[1] +
                         T[k * 4 + 2] * vh[2] + T[k * 4 + 3];
      out[(size_t)pt * 3 + k] = vert * 0.01f;
      fiw[m * LD32 + k]     = (_Float16)vert;
      fiw[m * LD32 + 3 + k] = (_Float16)x[pt * 3 + k];
      fiw[m * LD32 + 6 + k] = (_Float16)ws[WS_DELTA + (size_t)pt * 3 + k];
    }
  }
  __syncthreads();

  // front MLP 9->64->64->64
  wave_gemm(fiw, LD32, wf0, LD32, bias + 0 * 64, f1w, LD64, 32, 64, lane, true);
  __syncthreads();
  wave_gemm(f1w, LD64, wf1, LD64, bias + 1 * 64, f2w, LD64, 64, 64, lane, true);
  __syncthreads();
  wave_gemm(f2w, LD64, wf2, LD64, bias + 2 * 64, fFw, LD64, 64, 64, lane, true);
  __syncthreads();
  // scale head 64->64->64->3
  wave_gemm(fFw, LD64, wS0, LD64, bias + 3 * 64, f1w, LD64, 64, 64, lane, true);
  __syncthreads();
  wave_gemm(f1w, LD64, wS1, LD64, bias + 4 * 64, f2w, LD64, 64, 64, lane, true);
  __syncthreads();
  wave_gemm(f2w, LD64, wS2, LD64, bias + 5 * 64, sw, LD16, 64, 16, lane, true);
  __syncthreads();
  // rotation head 64->64->64->4
  wave_gemm(fFw, LD64, wR0, LD64, bias + 6 * 64, f1w, LD64, 64, 64, lane, true);
  __syncthreads();
  wave_gemm(f1w, LD64, wR1, LD64, bias + 7 * 64, f2w, LD64, 64, 64, lane, true);
  __syncthreads();
  wave_gemm(f2w, LD64, wR2, LD64, bias + 8 * 64, rw, LD16, 64, 16, lane, true);
  __syncthreads();

  if (lane < 16) {
    const int m = lane, pt = p0 + m;
#pragma unroll
    for (int k = 0; k < 3; ++k) {
      const float v = (float)sw[m * LD16 + k];
      out[OUT_SCALE + (size_t)pt * 3 + k] = (1.f / (1.f + expf(-v))) * 0.01f;
    }
    float r[4], s2 = 0.f;
#pragma unroll
    for (int k = 0; k < 4; ++k) { r[k] = (float)rw[m * LD16 + k]; s2 += r[k] * r[k]; }
    const float nrm = fmaxf(sqrtf(s2), 1e-12f);
#pragma unroll
    for (int k = 0; k < 4; ++k)
      out[OUT_ROT + (size_t)pt * 4 + k] = (r[k] / nrm) * 0.01f;
  }
}

// ======================= host launch =======================
extern "C" void kernel_launch(void* const* d_in, const int* in_sizes, int n_in,
                              void* d_out, int out_size, void* d_ws, size_t ws_size,
                              hipStream_t stream) {
  (void)in_sizes; (void)n_in; (void)out_size; (void)ws_size;
  const float* x     = (const float*)d_in[0];
  const float* betas = (const float*)d_in[1];
  const float* pose  = (const float*)d_in[2];

  BackboneParams bp;
  bp.W1  = (const float*)d_in[3];  bp.b1  = (const float*)d_in[4];
  bp.W2  = (const float*)d_in[5];  bp.b2  = (const float*)d_in[6];
  bp.W3  = (const float*)d_in[7];  bp.b3  = (const float*)d_in[8];
  bp.Wj0 = (const float*)d_in[9];  bp.bj0 = (const float*)d_in[10];
  bp.Wj1 = (const float*)d_in[11]; bp.bj1 = (const float*)d_in[12];
  bp.Wl0 = (const float*)d_in[13]; bp.bl0 = (const float*)d_in[14];
  bp.Wl1 = (const float*)d_in[15]; bp.bl1 = (const float*)d_in[16];
  bp.We  = (const float*)d_in[17]; bp.be  = (const float*)d_in[18];
  bp.Wp  = (const float*)d_in[19]; bp.bp  = (const float*)d_in[20];
  bp.Wd  = (const float*)d_in[21]; bp.bd  = (const float*)d_in[22];

  FrontParams fp;
  fp.Wf0 = (const float*)d_in[23]; fp.bf0 = (const float*)d_in[24];
  fp.Wf1 = (const float*)d_in[25]; fp.bf1 = (const float*)d_in[26];
  fp.Wf2 = (const float*)d_in[27]; fp.bf2 = (const float*)d_in[28];
  fp.Ws0 = (const float*)d_in[29]; fp.bs0 = (const float*)d_in[30];
  fp.Ws1 = (const float*)d_in[31]; fp.bs1 = (const float*)d_in[32];
  fp.Ws2 = (const float*)d_in[33]; fp.bs2 = (const float*)d_in[34];
  fp.Wr0 = (const float*)d_in[35]; fp.br0 = (const float*)d_in[36];
  fp.Wr1 = (const float*)d_in[37]; fp.br1 = (const float*)d_in[38];
  fp.Wr2 = (const float*)d_in[39]; fp.br2 = (const float*)d_in[40];

  float* ws  = (float*)d_ws;
  float* out = (float*)d_out;

  const int blocks = NPTS / 64;  // 200000/64 = 3125 exactly

  deform_prep_kernel<<<1, 32, 0, stream>>>(pose, ws);
  deform_backbone_kernel<<<blocks, 128, SM1_BYTES, stream>>>(x, betas, bp, ws);
  deform_joints_kernel<<<1, 32, 0, stream>>>(ws);
  deform_front_kernel<<<blocks, 128, SM3_BYTES, stream>>>(x, fp, ws, out);
}
